// SelfAttention_584115553100
// MI455X (gfx1250) — matrix operand order repaired and checked by reading, not compiled, not run on hardware
//
#include <hip/hip_runtime.h>

// ---- CDNA5 WMMA / TDM types -------------------------------------------------
typedef __attribute__((ext_vector_type(16))) __bf16          v16bf;
typedef __attribute__((ext_vector_type(8)))  float           v8f;
typedef __attribute__((ext_vector_type(16))) short           v16s;
typedef __attribute__((ext_vector_type(8)))  unsigned short  u16x8;
typedef __attribute__((ext_vector_type(4)))  unsigned int    u32x4;
typedef __attribute__((ext_vector_type(8)))  int             i32x8;
typedef __attribute__((ext_vector_type(4)))  int             i32x4;

#define WMMA_BF16(A, B, C) \
  __builtin_amdgcn_wmma_f32_16x16x32_bf16(false, (A), false, (B), (short)0, (C), false, false)

// ds_swizzle cross-lane xor (group-of-32 mode): imm = xor<<10 | 0x1F
#define SWZ_XOR(v, imm) \
  __int_as_float(__builtin_amdgcn_ds_swizzle(__float_as_int(v), (imm)))

__device__ __forceinline__ unsigned short f2bf(float f) {
  unsigned u = __float_as_uint(f);
  u = (u + 0x7FFFu + ((u >> 16) & 1u)) >> 16;   // RNE
  return (unsigned short)u;
}

// B-operand (K=32 x N=16 bf16): lane holds col n = lane%16,
// elems 0..7 -> K = hi*8 + e, elems 8..15 -> K = 16 + hi*8 + e   (p = base + hi*8)
__device__ __forceinline__ v16bf load_bop(const unsigned short* p) {
  u16x8 a = *(const u16x8*)p;
  u16x8 b = *(const u16x8*)(p + 16);
  v16s t;
#pragma unroll
  for (int e = 0; e < 8; ++e) { t[e] = (short)a[e]; t[8 + e] = (short)b[e]; }
  return __builtin_bit_cast(v16bf, t);
}

// A-operand (M=16 x K=32 bf16): lane holds row m = lane%16,
// elems 0..7 -> K = hi*8 + e, elems 8..15 -> K = 16 + hi*8 + e   (hi = lane/16)
__device__ __forceinline__ v16bf load_aop(const unsigned short* row, int hi) {
  u16x8 a = *(const u16x8*)(row + hi * 8);
  u16x8 b = *(const u16x8*)(row + 16 + hi * 8);
  v16s t;
#pragma unroll
  for (int e = 0; e < 8; ++e) { t[e] = (short)a[e]; t[8 + e] = (short)b[e]; }
  return __builtin_bit_cast(v16bf, t);
}

// ---- Tensor Data Mover: DMA a 32(pix) x 256(chan) bf16 tile of ht into LDS --
// ht tensor: [256 rows(c)] x [4096 cols(p)], element 2B. dim0 = p (contiguous),
// dim1 = c. LDS padding: +4 DWORDs every 16 DWORDs -> 80B row pitch
// (bank-conflict-free ds_load_b128 across 16 lanes).
__device__ __forceinline__ void tdm_load_ht(unsigned lds_off,
                                            unsigned long long gaddr) {
  u32x4 g0;
  g0[0] = 1u;                                            // count=1, no gather
  g0[1] = lds_off;                                       // lds_addr
  g0[2] = (unsigned)(gaddr & 0xFFFFFFFFu);               // global_addr[31:0]
  g0[3] = (unsigned)((gaddr >> 32) & 0x1FFFFFFu) | (2u << 30); // [56:32] | type=2
  i32x8 g1;
  g1[0] = (int)((1u << 16) |            // data_size = 2 bytes
                (1u << 20) |            // pad_enable
                (3u << 22) |            // pad_interval: every 16 DWORDs
                (3u << 25));            // pad_amount: 4 DWORDs
  g1[1] = (int)(4096u << 16);           // tensor_dim0 = 4096   (bits 63:48)
  g1[2] = (int)(256u  << 16);           // tensor_dim1 = 256    (bits 95:80)
  g1[3] = (int)(32u   << 16);           // tile_dim0   = 32     (bits 127:112)
  g1[4] = (int)(256u);                  // tile_dim1   = 256, tile_dim2 = 0
  g1[5] = (int)(4096u);                 // tensor_dim0_stride = 4096
  g1[6] = 0;                            // stride hi / tensor_dim1_stride lo
  g1[7] = 0;
  i32x4 z4 = {0, 0, 0, 0};
#if __clang_major__ >= 23
  i32x8 z8 = {0, 0, 0, 0, 0, 0, 0, 0};
  __builtin_amdgcn_tensor_load_to_lds(g0, g1, z4, z4, z8, 0);
#else
  __builtin_amdgcn_tensor_load_to_lds(g0, g1, z4, z4, 0);
#endif
}

// ---- Kernel 1: fused 1x1-conv projections -----------------------------------
// Logical GEMM [16384 x 320] = x[16384 x 256] * [Kf | Kg | Kh], one 16x16 tile
// per wave. f stored bf16 [pix][32]; g stored bf16 [pix][32] PRE-SCALED by
// log2(e) (softmax runs in exp2 domain); h stored bf16 transposed ht[b][c][pix].
__global__ __launch_bounds__(256) void proj_kernel(
    const float* __restrict__ x,
    const float* __restrict__ kf, const float* __restrict__ bf_,
    const float* __restrict__ kg, const float* __restrict__ bg_,
    const float* __restrict__ kh, const float* __restrict__ bh_,
    unsigned short* __restrict__ f_ws, unsigned short* __restrict__ g_ws,
    unsigned short* __restrict__ ht_ws)
{
  const int lane = threadIdx.x & 31;
  const int wav  = threadIdx.x >> 5;
  const int w    = blockIdx.x * 8 + wav;      // 20480 waves total
  const int nt   = w % 20;                    // N-tile: 0-1 f, 2-3 g, 4-19 h
  const int mt   = w / 20;                    // M-tile (1024)
  const int lm   = lane & 15;
  const int hi   = lane >> 4;

  const float* Wt; const float* bias; int ldw, col0, dst;
  if (nt < 2)      { Wt = kf; bias = bf_; ldw = 32;  col0 = nt * 16;        dst = 0; }
  else if (nt < 4) { Wt = kg; bias = bg_; ldw = 32;  col0 = (nt - 2) * 16;  dst = 1; }
  else             { Wt = kh; bias = bh_; ldw = 256; col0 = (nt - 4) * 16;  dst = 2; }

  const int    m    = mt * 16 + lm;           // A row (global pixel)
  const float* xr   = x + (size_t)m * 256;
  const int    gcol = col0 + lm;              // B / D column

  v8f acc = {0.f,0.f,0.f,0.f,0.f,0.f,0.f,0.f};
#pragma unroll
  for (int k0 = 0; k0 < 256; k0 += 32) {
    v16s av;
    {
      const float* pa = xr + k0 + hi * 16;
      const float* pb = xr + k0 + hi * 16 + 8;
#pragma unroll
      for (int e = 0; e < 8; ++e) { av[e] = (short)f2bf(pa[e]); av[8 + e] = (short)f2bf(pb[e]); }
    }
    v16s bv;
    {
      const float* pw = Wt + (size_t)(k0 + hi * 16) * ldw + gcol;
#pragma unroll
      for (int e = 0; e < 16; ++e) bv[e] = (short)f2bf(pw[(size_t)e * ldw]);
    }
    v16bf A = __builtin_bit_cast(v16bf, av);
    v16bf B = __builtin_bit_cast(v16bf, bv);
    acc = WMMA_BF16(A, B, acc);
  }
  const float bb  = bias[gcol];
  const float scl = (dst == 1) ? 1.44269504088896f : 1.0f;  // g in log2 domain
#pragma unroll
  for (int r = 0; r < 8; ++r) acc[r] = (acc[r] + bb) * scl;

  if (dst == 2) {                              // ht[b][c][pix], 8 contiguous pix
    const int mbase = mt * 16 + hi * 8;
    const int b  = mbase >> 12;
    const int p0 = mbase & 4095;
    u16x8 pk;
#pragma unroll
    for (int r = 0; r < 8; ++r) pk[r] = f2bf(acc[r]);
    *(u16x8*)(ht_ws + (((size_t)(b * 256 + gcol)) << 12) + p0) = pk;
  } else {
    unsigned short* o = (dst == 0) ? f_ws : g_ws;
#pragma unroll
    for (int r = 0; r < 8; ++r) {
      const int m2 = mt * 16 + hi * 8 + r;
      o[(size_t)m2 * 32 + gcol] = f2bf(acc[r]);
    }
  }
}

// ---- Kernel 2: fused flash attention + gamma residual -----------------------
// Block = 8 waves x 16 queries = 128 queries marching over keys in lockstep.
// TDM double-buffers the 32x256 ht slab into LDS; waves consume via ds_load.
#define HT_PITCH 40   // shorts per channel row in LDS: 32 data + 8 pad (80 B)

__global__ __launch_bounds__(256) void attn_kernel(
    const float* __restrict__ x, const float* __restrict__ gamma,
    const unsigned short* __restrict__ f_ws, const unsigned short* __restrict__ g_ws,
    const unsigned short* __restrict__ ht_ws, float* __restrict__ out)
{
  __shared__ unsigned short lds_ht[2][256 * HT_PITCH];   // 2 x 20 KB
  __shared__ float lds_l[8 * 16];

  const int lane = threadIdx.x & 31;
  const int wav  = threadIdx.x >> 5;
  const int lm   = lane & 15;
  const int hi   = lane >> 4;
  const int q0g  = blockIdx.x * 128 + wav * 16;   // global query-row base
  const int b    = q0g >> 12;

  // B operand for S^T: g^T (queries across lanes), already log2e-scaled
  const v16bf gB = load_bop(g_ws + (size_t)(q0g + lm) * 32 + hi * 8);

  float m_run = -3.0e38f, l_run = 0.0f;
  v8f oacc[16];
#pragma unroll
  for (int t = 0; t < 16; ++t) oacc[t] = (v8f){0.f,0.f,0.f,0.f,0.f,0.f,0.f,0.f};

  const unsigned short* fbase = f_ws + (((size_t)b) << 12) * 32;
  const unsigned short* hbase = ht_ws + (((size_t)b * 256) << 12);
  const unsigned lds0 = (unsigned)(size_t)&lds_ht[0][0];
  const unsigned lds1 = (unsigned)(size_t)&lds_ht[1][0];

  if (wav == 0)                                    // prologue DMA: j-block 0
    tdm_load_ht(lds0, (unsigned long long)(size_t)hbase);

  for (int it = 0; it < 128; ++it) {
    const int cur = it & 1;
    if (wav == 0) {
      if (it + 1 < 128) {                          // prefetch next slab via TDM
        tdm_load_ht(cur ? lds0 : lds1,
                    (unsigned long long)(size_t)(hbase + (it + 1) * 32));
        __builtin_amdgcn_s_wait_tensorcnt(1);      // older (current) slab done
      } else {
        __builtin_amdgcn_s_wait_tensorcnt(0);
      }
    }
    __syncthreads();                               // slab[cur] visible to all

    const int j0 = it * 32;
    const unsigned short* fr = fbase + (size_t)(j0 + lm) * 32;
    __builtin_prefetch(fr + 32 * 32, 0, 3);
    const v16bf A0 = load_aop(fr, hi);
    const v16bf A1 = load_aop(fr + 16 * 32, hi);
    const v8f z = {0.f,0.f,0.f,0.f,0.f,0.f,0.f,0.f};
    v8f st0 = WMMA_BF16(A0, gB, z);                // scores in log2 domain
    v8f st1 = WMMA_BF16(A1, gB, z);

    // wave-uniform running max
    float tmax = st0[0];
#pragma unroll
    for (int r = 1; r < 8; ++r) tmax = fmaxf(tmax, st0[r]);
#pragma unroll
    for (int r = 0; r < 8; ++r) tmax = fmaxf(tmax, st1[r]);
    tmax = fmaxf(tmax, SWZ_XOR(tmax, 0x041F));
    tmax = fmaxf(tmax, SWZ_XOR(tmax, 0x081F));
    tmax = fmaxf(tmax, SWZ_XOR(tmax, 0x101F));
    tmax = fmaxf(tmax, SWZ_XOR(tmax, 0x201F));
    tmax = fmaxf(tmax, SWZ_XOR(tmax, 0x401F));

    const float m_new = fmaxf(m_run, tmax);
    if (m_new > m_run) {                           // wave-uniform, rarely taken
      const float scale = exp2f(m_run - m_new);
      l_run *= scale;
#pragma unroll
      for (int t = 0; t < 16; ++t)
#pragma unroll
        for (int r = 0; r < 8; ++r) oacc[t][r] *= scale;
      m_run = m_new;
    }

    v16s pv; float lsum = 0.f;
#pragma unroll
    for (int r = 0; r < 8; ++r) {
      const float p = exp2f(st0[r] - m_run);       // v_exp_f32, no pre-mul
      lsum += p; pv[r] = (short)f2bf(p);
    }
#pragma unroll
    for (int r = 0; r < 8; ++r) {
      const float p = exp2f(st1[r] - m_run);
      lsum += p; pv[8 + r] = (short)f2bf(p);
    }
    l_run += lsum;
    const v16bf P = __builtin_bit_cast(v16bf, pv); // in-lane D -> A-operand map

    const unsigned short* hrow = &lds_ht[cur][lm * HT_PITCH + hi * 8];
#pragma unroll
    for (int ct = 0; ct < 16; ++ct) {
      const v16bf hB = load_bop(hrow + ct * 16 * HT_PITCH);  // ds_load_b128 x2
      oacc[ct] = WMMA_BF16(P, hB, oacc[ct]);
    }
    __syncthreads();                               // done with slab[cur]
  }

  // combine lane-pair partial sums (each half saw a disjoint half of the keys)
  const float l_tot = l_run + SWZ_XOR(l_run, 0x401F);
  if (lane < 16) lds_l[wav * 16 + lm] = l_tot;
  __syncthreads();

  const float gam = gamma[0];
  float rinv[8];
#pragma unroll
  for (int r = 0; r < 8; ++r) rinv[r] = gam / lds_l[wav * 16 + hi * 8 + r];

#pragma unroll
  for (int ct = 0; ct < 16; ++ct) {
#pragma unroll
    for (int r = 0; r < 8; ++r) {
      const int q = q0g + hi * 8 + r;              // query row of this D element
      const int c = ct * 16 + lm;                  // channel column
      const size_t idx = (size_t)q * 256 + c;
      out[idx] = rinv[r] * oacc[ct][r] + x[idx];
    }
  }
}

// ---- Launch -----------------------------------------------------------------
extern "C" void kernel_launch(void* const* d_in, const int* in_sizes, int n_in,
                              void* d_out, int out_size, void* d_ws, size_t ws_size,
                              hipStream_t stream) {
  const float* x     = (const float*)d_in[0];
  const float* kf    = (const float*)d_in[1];
  const float* bf_   = (const float*)d_in[2];
  const float* kg    = (const float*)d_in[3];
  const float* bg_   = (const float*)d_in[4];
  const float* kh    = (const float*)d_in[5];
  const float* bh_   = (const float*)d_in[6];
  const float* gamma = (const float*)d_in[7];
  float* out = (float*)d_out;

  unsigned short* f_ws  = (unsigned short*)d_ws;          // 16384*32 bf16
  unsigned short* g_ws  = f_ws + 16384 * 32;              // 16384*32 bf16
  unsigned short* ht_ws = g_ws + 16384 * 32;              // 4*256*4096 bf16

  // 20480 waves (1024 M-tiles x 20 N-tiles), 8 waves/block
  proj_kernel<<<2560, 256, 0, stream>>>(x, kf, bf_, kg, bg_, kh, bh_,
                                        f_ws, g_ws, ht_ws);
  // 128 blocks x 128 queries, TDM-fed flash attention
  attn_kernel<<<128, 256, 0, stream>>>(x, gamma, f_ws, g_ws, ht_ws, out);
}
